// attentionb_35390530519833
// MI455X (gfx1250) — compile-verified, matrix operand order checked by reading, not run on hardware
//
#include <hip/hip_runtime.h>
#include <hip/hip_bf16.h>
#include <cmath>

#define DIMS 512
#define HEADS 8
#define HD 64
#define BATCH 2
#define CTX 1536
#define NWIN 24
#define WIN 64
#define ROWS (BATCH * CTX)                 // 3072
#define NTOT (BATCH * HEADS * CTX * HD)    // 1572864

typedef _Float16 f16;
typedef __attribute__((ext_vector_type(16))) _Float16 v16h;
typedef __attribute__((ext_vector_type(8)))  _Float16 v8h;
typedef __attribute__((ext_vector_type(8)))  float    v8f;

// ---------------- swizzled operand layout ----------------
// WMMA f16 operand lane striping wants K runs {0..7,16..23} (lanes 0-15) and
// {8..15,24..31} (lanes 16-31).  kperm reorders a 32-K block so each lane's 16
// elements are CONTIGUOUS 32 bytes:  [k0..7][k16..23][k8..15][k24..31].
__device__ __forceinline__ int kperm(int k) {
  return ((k & 8) << 1) | ((k & 16) >> 1) | (k & 7);
}
// Operand tile stored as: base + (k>>5)*blockStride + row*32 + kperm(k&31)
// (row = M for A-operands, N for B-operands). Fragment load = one 32B read.
__device__ __forceinline__ v16h ld_swz(const f16* p, size_t blockStride, int row0, int k0) {
  int lane = threadIdx.x & 31;
  const f16* q = p + (size_t)(k0 >> 5) * blockStride +
                 (size_t)(row0 + (lane & 15)) * 32 + ((lane >> 4) << 4);
  return *(const v16h*)q;
}
// store 16 consecutive-k values (k0 multiple of 16) for one row into swizzled tile
__device__ __forceinline__ void st_swz16(f16* p, size_t blockStride, int r, int k0, v16h val) {
  int kk = k0 & 31;                       // 0 or 16
  f16* b2 = p + (size_t)(k0 >> 5) * blockStride + (size_t)r * 32 + (kk >> 1);
  *(v8h*)b2        = __builtin_shufflevector(val, val, 0, 1, 2, 3, 4, 5, 6, 7);
  *(v8h*)(b2 + 16) = __builtin_shufflevector(val, val, 8, 9, 10, 11, 12, 13, 14, 15);
}
__device__ __forceinline__ v8f wmma32(v16h a, v16h b, v8f c) {
  return __builtin_amdgcn_wmma_f32_16x16x32_f16(false, a, false, b, (short)0, c, false, false);
}

// ---------------- utility kernels ----------------
__global__ __launch_bounds__(256) void k_zero(float* p, int n) {
  int i = blockIdx.x * 256 + threadIdx.x;
  if (i < n) p[i] = 0.0f;
}

// f32 weight (K x N row-major) -> f16 swizzled-B global layout
__global__ __launch_bounds__(256) void k_cvt_swB(const float* __restrict__ W,
                                                 f16* __restrict__ out, int K, int N) {
  int i = blockIdx.x * 256 + threadIdx.x;
  if (i >= K * N) return;
  int k = i / N, n = i % N;
  out[(size_t)(k >> 5) * ((size_t)N * 32) + (size_t)n * 32 + kperm(k & 31)] = (f16)W[i];
}

// LayerNorm over last dim (512); output f16 in swizzled-A layout (rows = ROWS).
__global__ __launch_bounds__(256) void k_ln512(const float* __restrict__ x,
                                               const float* __restrict__ g,
                                               const float* __restrict__ bt,
                                               f16* __restrict__ out) {
  __shared__ float rs[256], rq[256];
  int row = blockIdx.x;
  const float* xr = x + (size_t)row * DIMS;
  int c = 2 * threadIdx.x;
  float v0 = xr[c], v1 = xr[c + 1];
  rs[threadIdx.x] = v0 + v1;
  rq[threadIdx.x] = v0 * v0 + v1 * v1;
  __syncthreads();
  for (int s = 128; s > 0; s >>= 1) {
    if (threadIdx.x < s) { rs[threadIdx.x] += rs[threadIdx.x + s]; rq[threadIdx.x] += rq[threadIdx.x + s]; }
    __syncthreads();
  }
  float mu = rs[0] * (1.0f / DIMS);
  float var = rq[0] * (1.0f / DIMS) - mu * mu;
  float r = rsqrtf(var + 1e-5f);
  size_t a0 = (size_t)(c >> 5) * ((size_t)ROWS * 32) + (size_t)row * 32 + kperm(c & 31);
  out[a0]     = (f16)((v0 - mu) * r * g[c] + bt[c]);
  out[a0 + 1] = (f16)((v1 - mu) * r * g[c + 1] + bt[c + 1]);  // c even -> adjacent under kperm
}

// C[M,N] f32 = A(swz, M rows) x B(swz, N cols).  LDS-free: fragments load as
// coalesced 32B global reads; weight/activation reuse lives in WGP$/L2.
__global__ __launch_bounds__(256) void k_gemm(const f16* __restrict__ A,
                                              const f16* __restrict__ Bm,
                                              float* __restrict__ C,
                                              int M, int N, int K) {
  int n0 = blockIdx.x * 64, m0 = blockIdx.y * 64;
  int wave = threadIdx.x >> 5, lane = threadIdx.x & 31;
  int mi = wave >> 1, nb = (wave & 1) * 2;
  size_t strA = (size_t)M * 32, strB = (size_t)N * 32;
  v8f acc0 = {}, acc1 = {};
  for (int k0 = 0; k0 < K; k0 += 32) {
    v16h a = ld_swz(A, strA, m0 + mi * 16, k0);
    acc0 = wmma32(a, ld_swz(Bm, strB, n0 + nb * 16, k0), acc0);
    acc1 = wmma32(a, ld_swz(Bm, strB, n0 + (nb + 1) * 16, k0), acc1);
  }
  int col = lane & 15, rb = (lane >> 4) << 3;
#pragma unroll
  for (int r = 0; r < 8; ++r) {
    int row = m0 + mi * 16 + rb + r;
    C[(size_t)row * N + n0 + nb * 16 + col]       = acc0[r];
    C[(size_t)row * N + n0 + (nb + 1) * 16 + col] = acc1[r];
  }
}

// Head split + RoPE + window gather for k/v. One thread per (g,c,pair).
__global__ __launch_bounds__(256) void k_heads_rope(const float* __restrict__ q_all,
                                                    const float* __restrict__ kv_all,
                                                    float* __restrict__ q,
                                                    float* __restrict__ k,
                                                    float* __restrict__ v,
                                                    const float* __restrict__ theta_p,
                                                    int cq, int window_mode) {
  int p = blockIdx.x * 256 + threadIdx.x;
  if (p >= NTOT / 2) return;
  int j = p & 31;
  int t = p >> 5;
  int c = t % cq;
  int g = t / cq;
  int h = g % HEADS;
  int b = (g / HEADS) % BATCH;
  int f = g / (HEADS * BATCH);
  float theta = theta_p[0];
  float fbj = 0.2f * (powf(21.0f, (float)j * (1.0f / 31.0f)) - 1.0f);  // mel freq base
  float ang = (float)c * (theta * (1.0f / 220.0f) * fbj);
  float cs = cosf(ang), sn = sinf(ang);
  int qrow = b * CTX + f * WIN + c;
  int kst  = window_mode ? max(0, f * WIN - (WIN / HEADS)) : 0;
  int krow = b * CTX + kst + c;
  int col  = h * HD + 2 * j;
  size_t o = (size_t)(g * cq + c) * HD + 2 * j;
  float qr = q_all[(size_t)qrow * DIMS + col];
  float qi = q_all[(size_t)qrow * DIMS + col + 1];
  q[o]     = qr * cs - qi * sn;
  q[o + 1] = qr * sn + qi * cs;
  float kr = kv_all[(size_t)krow * (2 * DIMS) + col];
  float ki = kv_all[(size_t)krow * (2 * DIMS) + col + 1];
  k[o]     = kr * cs - ki * sn;
  k[o + 1] = kr * sn + ki * cs;
  v[o]     = kv_all[(size_t)krow * (2 * DIMS) + DIMS + col];
  v[o + 1] = kv_all[(size_t)krow * (2 * DIMS) + DIMS + col + 1];
}

// Per-head GEMM: out = X(g,c,:) @ W(64x64) + bias; optional LN.
// do_ln=1: f16 out in swizzled-A layout (rowsTot rows)   [lnqt / lnkt]
// do_ln=0: f16 out in per-g key-swizzled layout           [vt for P@V]
__global__ __launch_bounds__(256) void k_head_gemm(const float* __restrict__ X,
                                                   const f16* __restrict__ Wsw,
                                                   const float* __restrict__ bias,
                                                   float* __restrict__ out32,
                                                   f16* __restrict__ out16,
                                                   int do_ln,
                                                   const float* __restrict__ lng,
                                                   const float* __restrict__ lnb,
                                                   int cq, int rowsTot) {
  __shared__ alignas(32) f16 Xs[64 * 64];
  __shared__ float Ts[64 * 64];
  __shared__ float redA[256], redB[256];
  __shared__ float mu[64], rstd[64];
  int g = blockIdx.y, c0 = blockIdx.x * 64;
  int tid = threadIdx.x;
  const float* Xg = X + ((size_t)g * cq + c0) * HD;
  {
    int rr = tid >> 2, cc0 = (tid & 3) * 16;
    const float4* src = (const float4*)(Xg + rr * 64 + cc0);
    v16h xv;
#pragma unroll
    for (int q4 = 0; q4 < 4; ++q4) {
      float4 fv = src[q4];
      xv[q4 * 4 + 0] = (f16)fv.x; xv[q4 * 4 + 1] = (f16)fv.y;
      xv[q4 * 4 + 2] = (f16)fv.z; xv[q4 * 4 + 3] = (f16)fv.w;
    }
    st_swz16(Xs, 64 * 32, rr, cc0, xv);
  }
  __syncthreads();
  int wave = tid >> 5, lane = tid & 31;
  int mi = wave >> 1, nb = (wave & 1) * 2;
  v8f acc0 = {}, acc1 = {};
#pragma unroll
  for (int k0 = 0; k0 < 64; k0 += 32) {
    v16h a = ld_swz(Xs, 64 * 32, mi * 16, k0);
    acc0 = wmma32(a, ld_swz(Wsw, 64 * 32, nb * 16, k0), acc0);
    acc1 = wmma32(a, ld_swz(Wsw, 64 * 32, (nb + 1) * 16, k0), acc1);
  }
  int col = lane & 15, rb = (lane >> 4) << 3;
#pragma unroll
  for (int r = 0; r < 8; ++r) {
    int row = mi * 16 + rb + r;
    float v0 = acc0[r] + bias[nb * 16 + col];
    float v1 = acc1[r] + bias[(nb + 1) * 16 + col];
    Ts[row * 64 + nb * 16 + col]       = v0;
    Ts[row * 64 + (nb + 1) * 16 + col] = v1;
    if (out32) {
      out32[((size_t)g * cq + c0 + row) * HD + nb * 16 + col]       = v0;
      out32[((size_t)g * cq + c0 + row) * HD + (nb + 1) * 16 + col] = v1;
    }
  }
  __syncthreads();
  if (do_ln) {
    int row = tid >> 2, sub = tid & 3;
    float s = 0.0f, sq = 0.0f;
#pragma unroll
    for (int i = 0; i < 16; ++i) { float xx = Ts[row * 64 + sub * 16 + i]; s += xx; sq += xx * xx; }
    redA[tid] = s; redB[tid] = sq;
    __syncthreads();
    if (sub == 0) {
      float ss = redA[tid] + redA[tid + 1] + redA[tid + 2] + redA[tid + 3];
      float qq = redB[tid] + redB[tid + 1] + redB[tid + 2] + redB[tid + 3];
      float m = ss * (1.0f / HD);
      mu[row] = m;
      rstd[row] = rsqrtf(qq * (1.0f / HD) - m * m + 1e-5f);
    }
    __syncthreads();
    int rr = tid >> 2, cc0 = (tid & 3) * 16;
    float m = mu[rr], rsd = rstd[rr];
    v16h ov;
#pragma unroll
    for (int i = 0; i < 16; ++i) {
      int cc = cc0 + i;
      ov[i] = (f16)((Ts[rr * 64 + cc] - m) * rsd * lng[cc] + lnb[cc]);
    }
    st_swz16(out16, (size_t)rowsTot * 32, g * cq + c0 + rr, cc0, ov);
  } else {
    // V: key-swizzled per g (row dim = d, k dim = key)
    int dd = tid >> 2, cc0 = (tid & 3) * 16;
    v16h ov;
#pragma unroll
    for (int i = 0; i < 16; ++i) ov[i] = (f16)Ts[(cc0 + i) * 64 + dd];
    st_swz16(out16 + (size_t)g * cq * HD, HD * 32, dd, c0 + cc0, ov);
  }
}

// Flash attention; Q/K in swizzled-A global layout (rowsTot rows, k = d),
// V in per-g key-swizzled layout. LDS only for softmax (Sf, Ps, row stats).
__global__ __launch_bounds__(256) void k_flash(const f16* __restrict__ Qsw,
                                               const f16* __restrict__ Ksw,
                                               const f16* __restrict__ Vsw,
                                               float* __restrict__ Out,
                                               int cq, int rowsTot, float scale) {
  __shared__ alignas(32) f16 Ps[64 * 64];
  __shared__ float Sf[64 * 64];
  __shared__ float mrow[64], lrow[64], arow[64];
  __shared__ float redA[256], redB[256];
  int g = blockIdx.y, q0 = blockIdx.x * 64;
  int tid = threadIdx.x;
  if (tid < 64) { mrow[tid] = -1e30f; lrow[tid] = 0.0f; }
  int wave = tid >> 5, lane = tid & 31;
  int mi = wave >> 1, nb = (wave & 1) * 2;
  int col = lane & 15, rb = (lane >> 4) << 3;
  int srow = tid >> 2, sub = tid & 3;
  size_t strR = (size_t)rowsTot * 32;
  const f16* Vg = Vsw + (size_t)g * cq * HD;
  // Q fragments resident in registers across the whole K loop
  v16h qf0 = ld_swz(Qsw, strR, g * cq + q0 + mi * 16, 0);
  v16h qf1 = ld_swz(Qsw, strR, g * cq + q0 + mi * 16, 32);
  v8f o0 = {}, o1 = {};
  int nkt = cq / 64;
  for (int kt = 0; kt < nkt; ++kt) {
    int kr = g * cq + kt * 64;
    v8f s0 = {}, s1 = {};
    s0 = wmma32(qf0, ld_swz(Ksw, strR, kr + nb * 16, 0), s0);
    s0 = wmma32(qf1, ld_swz(Ksw, strR, kr + nb * 16, 32), s0);
    s1 = wmma32(qf0, ld_swz(Ksw, strR, kr + (nb + 1) * 16, 0), s1);
    s1 = wmma32(qf1, ld_swz(Ksw, strR, kr + (nb + 1) * 16, 32), s1);
#pragma unroll
    for (int r = 0; r < 8; ++r) {
      int row = mi * 16 + rb + r;
      Sf[row * 64 + nb * 16 + col]       = s0[r] * scale;
      Sf[row * 64 + (nb + 1) * 16 + col] = s1[r] * scale;
    }
    __syncthreads();
    // online softmax: row max
    float mx = -1e30f;
#pragma unroll
    for (int i = 0; i < 16; ++i) mx = fmaxf(mx, Sf[srow * 64 + sub * 16 + i]);
    redA[tid] = mx;
    __syncthreads();
    if (sub == 0) {
      float m4 = fmaxf(fmaxf(redA[tid], redA[tid + 1]), fmaxf(redA[tid + 2], redA[tid + 3]));
      float mold = mrow[srow];
      float mnew = fmaxf(mold, m4);
      arow[srow] = __expf(mold - mnew);
      mrow[srow] = mnew;
    }
    __syncthreads();
    float mn = mrow[srow];
    float ps = 0.0f;
    v16h pv;
#pragma unroll
    for (int i = 0; i < 16; ++i) {
      float p = __expf(Sf[srow * 64 + sub * 16 + i] - mn);
      pv[i] = (f16)p;
      ps += p;
    }
    st_swz16(Ps, 64 * 32, srow, sub * 16, pv);   // P as swizzled A-operand
    redB[tid] = ps;
    __syncthreads();
    if (sub == 0)
      lrow[srow] = arow[srow] * lrow[srow] + redB[tid] + redB[tid + 1] + redB[tid + 2] + redB[tid + 3];
    __syncthreads();
    // O = O*alpha + P V
#pragma unroll
    for (int r = 0; r < 8; ++r) {
      float al = arow[mi * 16 + rb + r];
      o0[r] *= al; o1[r] *= al;
    }
#pragma unroll
    for (int kb = 0; kb < 64; kb += 32) {
      v16h a = ld_swz(Ps, 64 * 32, mi * 16, kb);
      o0 = wmma32(a, ld_swz(Vg, HD * 32, nb * 16, kt * 64 + kb), o0);
      o1 = wmma32(a, ld_swz(Vg, HD * 32, (nb + 1) * 16, kt * 64 + kb), o1);
    }
  }
  __syncthreads();
#pragma unroll
  for (int r = 0; r < 8; ++r) {
    int row = mi * 16 + rb + r;
    float inv = 1.0f / lrow[row];
    Out[((size_t)g * cq + q0 + row) * HD + nb * 16 + col]       = o0[r] * inv;
    Out[((size_t)g * cq + q0 + row) * HD + (nb + 1) * 16 + col] = o1[r] * inv;
  }
}

// sum |a - prev| per focus instance
__global__ __launch_bounds__(256) void k_absdiff(const float* __restrict__ a,
                                                 const float* __restrict__ pr,
                                                 float* __restrict__ dsum, int per_f) {
  __shared__ float red[256];
  int idx = blockIdx.x * 256 + threadIdx.x;
  red[threadIdx.x] = fabsf(a[idx] - pr[idx]);
  __syncthreads();
  for (int s = 128; s > 0; s >>= 1) {
    if (threadIdx.x < s) red[threadIdx.x] += red[threadIdx.x + s];
    __syncthreads();
  }
  if (threadIdx.x == 0) atomicAdd(&dsum[(blockIdx.x * 256) / per_f], red[0]);
}

// iteration state update (reference jnp.where logic); done[it] read, done[it+1] written
__global__ __launch_bounds__(256) void k_update(float* __restrict__ attn,
                                                float* __restrict__ prev,
                                                float* __restrict__ curq,
                                                float* __restrict__ k,
                                                float* __restrict__ v,
                                                const float* __restrict__ a,
                                                const float* __restrict__ kt,
                                                const float* __restrict__ vt,
                                                const float* __restrict__ dsum,
                                                float* __restrict__ done,
                                                int it, int F, int per_f, float inv_cnt) {
  int idx = blockIdx.x * 256 + threadIdx.x;
  int f = idx / per_f;
  float diff = dsum[f] * inv_cnt;
  bool stop = (it > 0) && (diff < 0.5f);
  bool dn = done[it * F + f] > 0.5f;
  float av = a[idx];
  if (!dn) {
    attn[idx] = av;
    k[idx] = kt[idx];
    v[idx] = vt[idx];
  }
  if (!dn && !stop) {
    prev[idx] = av;
    curq[idx] += av;
  }
  if (idx == f * per_f) done[(it + 1) * F + f] = (dn || stop) ? 1.0f : 0.0f;
}

// merge heads for slide stage: y[b, f*64+c, h*64+d] = attn[(((f*B+b)*H+h)*64+c)*64+d]
__global__ __launch_bounds__(256) void k_merge_slide(const float* __restrict__ attn,
                                                     float* __restrict__ y) {
  int idx = blockIdx.x * 256 + threadIdx.x;
  int d = idx % HD;
  int c = (idx / HD) % WIN;
  int g = idx / (WIN * HD);
  int h = g % HEADS;
  int b = (g / HEADS) % BATCH;
  int f = g / (HEADS * BATCH);
  y[(size_t)(b * CTX + f * WIN + c) * DIMS + h * HD + d] = attn[idx];
}

// merge heads for final focus; f16 swizzled-A output (rows = ROWS) for Wo GEMM
__global__ __launch_bounds__(256) void k_merge_final(const float* __restrict__ attn,
                                                     f16* __restrict__ out) {
  int idx = blockIdx.x * 256 + threadIdx.x;
  int d = idx % HD;
  int c = (idx / HD) % CTX;
  int g = idx / (CTX * HD);
  int h = g % HEADS;
  int b = g / HEADS;
  int row = b * CTX + c;
  int colc = h * HD + d;
  out[(size_t)(colc >> 5) * ((size_t)ROWS * 32) + (size_t)row * 32 + kperm(colc & 31)] =
      (f16)attn[idx];
}

// ---------------- host orchestration ----------------
struct FocusCtx {
  const float *lna_g, *lna_b, *lnb_g, *lnb_b, *bq, *bk, *bv, *theta;
  f16 *Wq_h, *Wkv_h, *Wqhd, *Wkhd, *Wvhd;
  f16 *lnA, *lnkt, *vth;       // lnA doubles as lnx (prep) / lnqt (iters)
  float *bufA, *bufB;          // bufA: q_all | kt ; bufB: kv_all | (vt, a)
  float *qbuf, *kbuf, *vbuf, *prevb, *attnb;
  float *dsum, *dones;
};

static void run_focus(const FocusCtx& C, const float* srcq, const float* srckv,
                      int F, int cq, int window_mode, hipStream_t s) {
  int gcount = F * BATCH * HEADS;
  int per_f = BATCH * HEADS * cq * HD;
  int rowsTot = gcount * cq;                 // == NTOT / HD
  float* q_all = C.bufA;
  float* kv_all = C.bufB;
  k_ln512<<<ROWS, 256, 0, s>>>(srcq, C.lna_g, C.lna_b, C.lnA);
  k_gemm<<<dim3(DIMS / 64, ROWS / 64), 256, 0, s>>>(C.lnA, C.Wq_h, q_all, ROWS, DIMS, DIMS);
  k_ln512<<<ROWS, 256, 0, s>>>(srckv, C.lna_g, C.lna_b, C.lnA);
  k_gemm<<<dim3(2 * DIMS / 64, ROWS / 64), 256, 0, s>>>(C.lnA, C.Wkv_h, kv_all, ROWS, 2 * DIMS, DIMS);
  k_heads_rope<<<(NTOT / 2) / 256, 256, 0, s>>>(q_all, kv_all, C.qbuf, C.kbuf, C.vbuf,
                                                C.theta, cq, window_mode);
  k_zero<<<NTOT / 256, 256, 0, s>>>(C.prevb, NTOT);
  k_zero<<<NTOT / 256, 256, 0, s>>>(C.attnb, NTOT);
  k_zero<<<1, 256, 0, s>>>(C.dones, 256);
  float* kt = C.bufA;
  float* vt = C.bufB;
  float* ab = C.bufB + NTOT;
  f16* lnqt = C.lnA;
  for (int it = 0; it < 3; ++it) {
    k_head_gemm<<<dim3(cq / 64, gcount), 256, 0, s>>>(C.qbuf, C.Wqhd, C.bq, nullptr, lnqt, 1,
                                                      C.lnb_g, C.lnb_b, cq, rowsTot);
    k_head_gemm<<<dim3(cq / 64, gcount), 256, 0, s>>>(C.kbuf, C.Wkhd, C.bk, kt, C.lnkt, 1,
                                                      C.lnb_g, C.lnb_b, cq, rowsTot);
    k_head_gemm<<<dim3(cq / 64, gcount), 256, 0, s>>>(C.vbuf, C.Wvhd, C.bv, vt, C.vth, 0,
                                                      C.lnb_g, C.lnb_b, cq, rowsTot);
    k_zero<<<1, 256, 0, s>>>(C.dsum, F);
    float temp = 1.0f - 0.005f * (float)it;
    float scale = 0.125f * ((it == 0) ? 1.0f : 1.0f / sqrtf(temp));
    k_flash<<<dim3(cq / 64, gcount), 256, 0, s>>>(lnqt, C.lnkt, C.vth, ab, cq, rowsTot, scale);
    k_absdiff<<<NTOT / 256, 256, 0, s>>>(ab, C.prevb, C.dsum, per_f);
    k_update<<<NTOT / 256, 256, 0, s>>>(C.attnb, C.prevb, C.qbuf, C.kbuf, C.vbuf, ab, kt, vt,
                                        C.dsum, C.dones, it, F, per_f, 1.0f / (float)per_f);
  }
}

extern "C" void kernel_launch(void* const* d_in, const int* in_sizes, int n_in,
                              void* d_out, int out_size, void* d_ws, size_t ws_size,
                              hipStream_t stream) {
  (void)in_sizes; (void)n_in; (void)out_size; (void)ws_size;
  const float* x     = (const float*)d_in[0];
  const float* xa    = (const float*)d_in[1];
  const float* Wq    = (const float*)d_in[2];
  const float* Wkv   = (const float*)d_in[3];
  const float* Wo    = (const float*)d_in[4];
  const float* lna_g = (const float*)d_in[5];
  const float* lna_b = (const float*)d_in[6];
  const float* lnb_g = (const float*)d_in[7];
  const float* lnb_b = (const float*)d_in[8];
  const float* Wq_hd = (const float*)d_in[9];
  const float* bq_hd = (const float*)d_in[10];
  const float* Wk_hd = (const float*)d_in[11];
  const float* bk_hd = (const float*)d_in[12];
  const float* Wv_hd = (const float*)d_in[13];
  const float* bv_hd = (const float*)d_in[14];
  const float* theta = (const float*)d_in[15];

  size_t off = 0;
  char* base = (char*)d_ws;
  auto alloc = [&](size_t bytes) -> void* {
    void* p = base + off;
    off += (bytes + 255) & ~(size_t)255;
    return p;
  };
  f16* Wq_h  = (f16*)alloc((size_t)DIMS * DIMS * 2);
  f16* Wkv_h = (f16*)alloc((size_t)DIMS * 2 * DIMS * 2);
  f16* Wo_h  = (f16*)alloc((size_t)DIMS * DIMS * 2);
  f16* Wqhd  = (f16*)alloc(HD * HD * 2);
  f16* Wkhd  = (f16*)alloc(HD * HD * 2);
  f16* Wvhd  = (f16*)alloc(HD * HD * 2);
  f16* lnA   = (f16*)alloc((size_t)ROWS * DIMS * 2);          // lnx / lnqt
  float* bufA = (float*)alloc((size_t)ROWS * DIMS * 4);       // q_all / kt
  float* bufB = (float*)alloc((size_t)ROWS * 2 * DIMS * 4);   // kv_all / (vt, a)
  float* qbuf  = (float*)alloc((size_t)NTOT * 4);             // q -> curq
  float* kbuf  = (float*)alloc((size_t)NTOT * 4);
  float* vbuf  = (float*)alloc((size_t)NTOT * 4);
  float* prevb = (float*)alloc((size_t)NTOT * 4);
  float* attnb = (float*)alloc((size_t)NTOT * 4);
  f16* lnkt = (f16*)alloc((size_t)NTOT * 2);
  f16* vth  = (f16*)alloc((size_t)NTOT * 2);
  float* ybuf  = (float*)alloc((size_t)NTOT * 4);
  float* yabuf = (float*)alloc((size_t)NTOT * 4);
  f16* finh = (f16*)alloc((size_t)NTOT * 2);
  float* dsum  = (float*)alloc(256 * 4);
  float* dones = (float*)alloc(256 * 4);

  // weight conversions -> f16 swizzled-B layouts
  k_cvt_swB<<<(DIMS * DIMS) / 256, 256, 0, stream>>>(Wq, Wq_h, DIMS, DIMS);
  k_cvt_swB<<<(DIMS * 2 * DIMS) / 256, 256, 0, stream>>>(Wkv, Wkv_h, DIMS, 2 * DIMS);
  k_cvt_swB<<<(DIMS * DIMS) / 256, 256, 0, stream>>>(Wo, Wo_h, DIMS, DIMS);
  k_cvt_swB<<<(HD * HD) / 256, 256, 0, stream>>>(Wq_hd, Wqhd, HD, HD);
  k_cvt_swB<<<(HD * HD) / 256, 256, 0, stream>>>(Wk_hd, Wkhd, HD, HD);
  k_cvt_swB<<<(HD * HD) / 256, 256, 0, stream>>>(Wv_hd, Wvhd, HD, HD);

  FocusCtx C;
  C.lna_g = lna_g; C.lna_b = lna_b; C.lnb_g = lnb_g; C.lnb_b = lnb_b;
  C.bq = bq_hd; C.bk = bk_hd; C.bv = bv_hd; C.theta = theta;
  C.Wq_h = Wq_h; C.Wkv_h = Wkv_h; C.Wqhd = Wqhd; C.Wkhd = Wkhd; C.Wvhd = Wvhd;
  C.lnA = lnA; C.lnkt = lnkt; C.vth = vth;
  C.bufA = bufA; C.bufB = bufB;
  C.qbuf = qbuf; C.kbuf = kbuf; C.vbuf = vbuf; C.prevb = prevb; C.attnb = attnb;
  C.dsum = dsum; C.dones = dones;

  // y = _slide(x)
  run_focus(C, x, x, NWIN, WIN, 1, stream);
  k_merge_slide<<<NTOT / 256, 256, 0, stream>>>(attnb, ybuf);
  // ya = _slide(xa)
  run_focus(C, xa, xa, NWIN, WIN, 1, stream);
  k_merge_slide<<<NTOT / 256, 256, 0, stream>>>(attnb, yabuf);
  // final: _focus(y, ya)
  run_focus(C, ybuf, yabuf, 1, CTX, 0, stream);
  k_merge_final<<<NTOT / 256, 256, 0, stream>>>(attnb, finh);
  // @ Wo
  k_gemm<<<dim3(DIMS / 64, ROWS / 64), 256, 0, stream>>>(finh, Wo_h, (float*)d_out,
                                                         ROWS, DIMS, DIMS);
}
